// GNN_84250078478617
// MI455X (gfx1250) — matrix operand order repaired and checked by reading, not compiled, not run on hardware
//
#include <hip/hip_runtime.h>
#include <math.h>

#define NN 100000
#define EE 1600000
#define DD 256
#define GG 50
#define LL 3

typedef __attribute__((ext_vector_type(16))) __bf16   v16bf;
typedef __attribute__((ext_vector_type(8)))  float    v8f;
typedef __attribute__((ext_vector_type(4)))  unsigned int u32x4;

union Frag { v16bf v; u32x4 u[2]; };

__device__ __forceinline__ unsigned short f2bf(float f) {
    unsigned int u = __float_as_uint(f);
    u += 0x7FFFu + ((u >> 16) & 1u);   // round-to-nearest-even
    return (unsigned short)(u >> 16);
}

__device__ __forceinline__ float gelu_exact(float x) {
    return 0.5f * x * (1.0f + erff(x * 0.70710678118654752440f));
}

// ---------------- utility ----------------
__global__ void k_zero(float* p, int n) {
    int i = blockIdx.x * blockDim.x + threadIdx.x;
    if (i < n) p[i] = 0.0f;
}

// ---------------- degree / counts ----------------
__global__ void k_deg(const long long* __restrict__ ei, float* __restrict__ deg) {
    int e = blockIdx.x * blockDim.x + threadIdx.x;
    if (e < EE) atomicAdd(&deg[(int)ei[EE + e]], 1.0f);
}

__global__ void k_cnt(const int* __restrict__ batch, float* __restrict__ cnt) {
    int i = blockIdx.x * blockDim.x + threadIdx.x;
    if (i < NN) atomicAdd(&cnt[batch[i]], 1.0f);
}

__global__ void k_dinv(const float* __restrict__ deg, float* __restrict__ dinv) {
    int i = blockIdx.x * blockDim.x + threadIdx.x;
    if (i < NN) dinv[i] = rsqrtf(1.0f + deg[i]);
}

// ---------------- bf16 cast & W fragment-order pack ----------------
__global__ void k_cast(const float* __restrict__ h, unsigned short* __restrict__ hb) {
    int i = blockIdx.x * blockDim.x + threadIdx.x;
    if (i < NN * DD) hb[i] = f2bf(h[i]);
}

// Pack W[l] (row-major KxN, K=N=256) into per-(col-tile, k-step, lane) contiguous
// 32B chunks holding B fragments in contiguous k order (k = 16*half + e), the order k_gemm loads A in.
__global__ void k_pack(const float* __restrict__ W, unsigned short* __restrict__ wp) {
    int idx = blockIdx.x * blockDim.x + threadIdx.x;
    if (idx >= LL * 16 * 8 * 32 * 16) return;
    int e    =  idx        & 15;
    int lane = (idx >> 4)  & 31;
    int ks   = (idx >> 9)  & 7;
    int ct   = (idx >> 12) & 15;
    int l    =  idx >> 16;
    int n    = ct * 16 + (lane & 15);
    int half = lane >> 4;
    int k    = ks * 32 + half * 16 + e;
    wp[idx] = f2bf(W[((size_t)l * DD + k) * DD + n]);
}

// ---------------- WMMA GEMM: out[N,256] = hb[N,256] @ Wb[256,256] ----------------
// Software-pipelined: fragments for k-step ks+1 are loaded before the WMMAs of
// step ks issue, so VMEM latency overlaps matrix math within one wave.
__global__ void __launch_bounds__(128) k_gemm(const unsigned short* __restrict__ hb,
                                              const unsigned short* __restrict__ wp,
                                              float* __restrict__ out) {
    const int lane = threadIdx.x & 31;
    const int wave = threadIdx.x >> 5;
    const int rt   = blockIdx.x;          // 16-row tile index
    const int m    = lane & 15;           // row (A) / col (B,C) within tile
    const int half = lane >> 4;
    const int row  = rt * 16 + m;
    const int ct0  = wave * 4;            // this wave covers col tiles ct0..ct0+3

    const unsigned short* aprow = hb + (size_t)row * DD + half * 16;
    const unsigned short* bbase = wp + (size_t)ct0 * 4096 + lane * 16;

    auto loadA = [&](Frag& f, int ks) {
        const unsigned short* p = aprow + ks * 32;
        f.u[0] = *(const u32x4*)(p);        // k = ks*32 + 16*half + {0..7}
        f.u[1] = *(const u32x4*)(p + 8);    // k = ks*32 + 16*half + 8 + {0..7}
    };
    auto loadB = [&](Frag& f, int c, int ks) {
        const unsigned short* p = bbase + c * 4096 + ks * 512;
        f.u[0] = *(const u32x4*)(p);
        f.u[1] = *(const u32x4*)(p + 8);
    };

    v8f acc0 = {}, acc1 = {}, acc2 = {}, acc3 = {};

    Frag a[2];
    Frag b[2][4];
    loadA(a[0], 0);
    loadB(b[0][0], 0, 0);
    loadB(b[0][1], 1, 0);
    loadB(b[0][2], 2, 0);
    loadB(b[0][3], 3, 0);

    #pragma unroll
    for (int ks = 0; ks < 8; ++ks) {
        const int cur = ks & 1;
        const int nxt = cur ^ 1;
        if (ks < 7) {
            loadA(a[nxt], ks + 1);
            loadB(b[nxt][0], 0, ks + 1);
            loadB(b[nxt][1], 1, ks + 1);
            loadB(b[nxt][2], 2, ks + 1);
            loadB(b[nxt][3], 3, ks + 1);
        }
        acc0 = __builtin_amdgcn_wmma_f32_16x16x32_bf16(false, a[cur].v, false, b[cur][0].v, (short)0, acc0, false, false);
        acc1 = __builtin_amdgcn_wmma_f32_16x16x32_bf16(false, a[cur].v, false, b[cur][1].v, (short)0, acc1, false, false);
        acc2 = __builtin_amdgcn_wmma_f32_16x16x32_bf16(false, a[cur].v, false, b[cur][2].v, (short)0, acc2, false, false);
        acc3 = __builtin_amdgcn_wmma_f32_16x16x32_bf16(false, a[cur].v, false, b[cur][3].v, (short)0, acc3, false, false);
    }

    #pragma unroll
    for (int v = 0; v < 8; ++v) {
        size_t r = (size_t)(rt * 16 + v + 8 * half) * DD;
        out[r + (ct0 + 0) * 16 + m] = acc0[v];
        out[r + (ct0 + 1) * 16 + m] = acc1[v];
        out[r + (ct0 + 2) * 16 + m] = acc2[v];
        out[r + (ct0 + 3) * 16 + m] = acc3[v];
    }
}

// ---------------- edge gather/scatter: agg[dst] += hw[src]*coef ----------------
__global__ void k_edge(const long long* __restrict__ ei, const float* __restrict__ dinv,
                       const float* __restrict__ hw, float* __restrict__ agg) {
    int idx = blockIdx.x * blockDim.x + threadIdx.x;   // EE*64 threads, 4 floats each
    if (idx >= EE * 64) return;
    int e = idx >> 6;
    int q = (idx & 63) << 2;
    long long s = ei[e];
    long long d = ei[EE + e];
    float coef = dinv[(int)s] * dinv[(int)d];
    float4 mv = *(const float4*)(hw + (size_t)s * DD + q);
    float* ap = agg + (size_t)d * DD + q;
    atomicAdd(ap + 0, mv.x * coef);
    atomicAdd(ap + 1, mv.y * coef);
    atomicAdd(ap + 2, mv.z * coef);
    atomicAdd(ap + 3, mv.w * coef);
}

// ---------------- fused: h = gelu(agg + hw*dinv^2 + b); per-graph sum/sumsq ----------------
__global__ void __launch_bounds__(64) k_combine(const float* __restrict__ hw, float* __restrict__ h,
                                                const float* __restrict__ dinv, const float* __restrict__ bias,
                                                const int* __restrict__ batch,
                                                float* __restrict__ gsum, float* __restrict__ gsq) {
    int node = blockIdx.x;
    int t = threadIdx.x;
    int d = t << 2;
    float di = dinv[node];
    float dsq = di * di;
    float4 w4 = *(const float4*)(hw + (size_t)node * DD + d);
    float4 a4 = *(const float4*)(h  + (size_t)node * DD + d);
    float4 b4 = *(const float4*)(bias + d);
    float gx = gelu_exact(a4.x + w4.x * dsq + b4.x);
    float gy = gelu_exact(a4.y + w4.y * dsq + b4.y);
    float gz = gelu_exact(a4.z + w4.z * dsq + b4.z);
    float gw = gelu_exact(a4.w + w4.w * dsq + b4.w);
    float4 o4 = { gx, gy, gz, gw };
    *(float4*)(h + (size_t)node * DD + d) = o4;

    __shared__ float ss[64], sq[64];
    ss[t] = gx + gy + gz + gw;
    sq[t] = gx * gx + gy * gy + gz * gz + gw * gw;
    __syncthreads();
    #pragma unroll
    for (int off = 32; off > 0; off >>= 1) {
        if (t < off) { ss[t] += ss[t + off]; sq[t] += sq[t + off]; }
        __syncthreads();
    }
    if (t == 0) {
        int g = batch[node];
        atomicAdd(&gsum[g], ss[0]);
        atomicAdd(&gsq[g],  sq[0]);
    }
}

__global__ void k_stats(const float* __restrict__ gsum, const float* __restrict__ gsq,
                        const float* __restrict__ cnt, float* __restrict__ mean, float* __restrict__ inv) {
    int g = blockIdx.x * blockDim.x + threadIdx.x;
    if (g < GG) {
        float c  = cnt[g] * (float)DD;
        float mu = gsum[g] / c;
        float e2 = gsq[g] / c;
        mean[g] = mu;
        inv[g]  = rsqrtf(e2 - mu * mu + 1e-5f);
    }
}

__global__ void k_norm(float* __restrict__ h, const int* __restrict__ batch,
                       const float* __restrict__ mean, const float* __restrict__ inv,
                       const float* __restrict__ gamma, const float* __restrict__ beta) {
    int i = blockIdx.x * blockDim.x + threadIdx.x;   // N*D/4
    if (i >= NN * DD / 4) return;
    int node = i >> 6;
    int d = (i & 63) << 2;
    int g = batch[node];
    float mu = mean[g], iv = inv[g];
    float4 x4  = *(float4*)(h + (size_t)node * DD + d);
    float4 g4  = *(const float4*)(gamma + d);
    float4 be4 = *(const float4*)(beta + d);
    x4.x = (x4.x - mu) * iv * g4.x + be4.x;
    x4.y = (x4.y - mu) * iv * g4.y + be4.y;
    x4.z = (x4.z - mu) * iv * g4.z + be4.z;
    x4.w = (x4.w - mu) * iv * g4.w + be4.w;
    *(float4*)(h + (size_t)node * DD + d) = x4;
}

// ---------------- residual + mean pool ----------------
__global__ void k_pool(const float* __restrict__ h, const float* __restrict__ x,
                       const int* __restrict__ batch, float* __restrict__ pooled) {
    int i = blockIdx.x * blockDim.x + threadIdx.x;   // N*D/4
    if (i >= NN * DD / 4) return;
    int node = i >> 6;
    int d = (i & 63) << 2;
    int g = batch[node];
    float4 hv = *(const float4*)(h + (size_t)node * DD + d);
    float4 xv = *(const float4*)(x + (size_t)node * DD + d);
    float* p = pooled + g * DD + d;
    atomicAdd(p + 0, hv.x + xv.x);
    atomicAdd(p + 1, hv.y + xv.y);
    atomicAdd(p + 2, hv.z + xv.z);
    atomicAdd(p + 3, hv.w + xv.w);
}

__global__ void k_out(const float* __restrict__ pooled, const float* __restrict__ cnt,
                      float* __restrict__ out) {
    int i = blockIdx.x * blockDim.x + threadIdx.x;
    if (i < GG * DD) out[i] = pooled[i] / cnt[i >> 8];
}

extern "C" void kernel_launch(void* const* d_in, const int* in_sizes, int n_in,
                              void* d_out, int out_size, void* d_ws, size_t ws_size,
                              hipStream_t stream) {
    const float*     x     = (const float*)d_in[0];
    const long long* ei    = (const long long*)d_in[1];   // (2, E) int64
    const int*       batch = (const int*)d_in[3];
    const float*     W     = (const float*)d_in[4];       // (L, D, D)
    const float*     bias  = (const float*)d_in[5];       // (L, D)
    const float*     gamma = (const float*)d_in[6];
    const float*     beta  = (const float*)d_in[7];
    float*           out   = (float*)d_out;               // (G, D)

    char* ws = (char*)d_ws;
    size_t off = 0;
    auto alloc = [&](size_t bytes) -> char* {
        char* p = ws + off;
        off = (off + bytes + 255) & ~(size_t)255;
        return p;
    };

    unsigned short* hb    = (unsigned short*)alloc((size_t)NN * DD * 2);         // bf16 h
    float*          bufW  = (float*)alloc((size_t)NN * DD * 4);                  // h @ W
    float*          bufH  = (float*)alloc((size_t)NN * DD * 4);                  // agg / h
    unsigned short* wpack = (unsigned short*)alloc((size_t)LL * 65536 * 2);      // swizzled W
    float*          deg   = (float*)alloc((size_t)NN * 4);
    float*          dinv  = (float*)alloc((size_t)NN * 4);
    float*          cnt   = (float*)alloc((size_t)GG * 4);
    float*          gsum  = (float*)alloc((size_t)GG * 4);
    float*          gsq   = (float*)alloc((size_t)GG * 4);
    float*          mean  = (float*)alloc((size_t)GG * 4);
    float*          inv   = (float*)alloc((size_t)GG * 4);
    float*          pooled= (float*)alloc((size_t)GG * DD * 4);

    k_zero<<<(NN + 255) / 256, 256, 0, stream>>>(deg, NN);
    k_zero<<<1, 64, 0, stream>>>(cnt, GG);
    k_deg<<<(EE + 255) / 256, 256, 0, stream>>>(ei, deg);
    k_cnt<<<(NN + 255) / 256, 256, 0, stream>>>(batch, cnt);
    k_dinv<<<(NN + 255) / 256, 256, 0, stream>>>(deg, dinv);
    k_pack<<<(LL * 65536 + 255) / 256, 256, 0, stream>>>(W, wpack);

    const float* hcur = x;
    for (int l = 0; l < LL; ++l) {
        k_cast<<<(NN * DD + 255) / 256, 256, 0, stream>>>(hcur, hb);
        k_zero<<<(NN * DD + 255) / 256, 256, 0, stream>>>(bufH, NN * DD);
        k_zero<<<1, 64, 0, stream>>>(gsum, GG);
        k_zero<<<1, 64, 0, stream>>>(gsq, GG);
        k_gemm<<<NN / 16, 128, 0, stream>>>(hb, wpack + (size_t)l * 65536, bufW);
        k_edge<<<(EE * 64) / 256, 256, 0, stream>>>(ei, dinv, bufW, bufH);
        k_combine<<<NN, 64, 0, stream>>>(bufW, bufH, dinv, bias + l * DD, batch, gsum, gsq);
        k_stats<<<1, 64, 0, stream>>>(gsum, gsq, cnt, mean, inv);
        k_norm<<<(NN * DD / 4 + 255) / 256, 256, 0, stream>>>(bufH, batch, mean, inv,
                                                              gamma + l * DD, beta + l * DD);
        hcur = bufH;
    }

    k_zero<<<(GG * DD + 255) / 256, 256, 0, stream>>>(pooled, GG * DD);
    k_pool<<<(NN * DD / 4 + 255) / 256, 256, 0, stream>>>(bufH, x, batch, pooled);
    k_out<<<(GG * DD + 255) / 256, 256, 0, stream>>>(pooled, cnt, out);
}
